// FastDiffuserAttention_24232205484115
// MI455X (gfx1250) — matrix operand order repaired and checked by reading, not compiled, not run on hardware
//
#include <hip/hip_runtime.h>

#define BB 4
#define SS 2048
#define HH 2048
#define NHD 16
#define NKV 8
#define HD 128
#define QSZ (NHD*HD)          // 2048
#define KVSZ (NKV*HD)         // 1024
#define OQKV (QSZ + 2*KVSZ)   // 4096
#define NTOK (BB*SS)          // 8192
#define SCL 0.08838834764831845f  // 128^-0.5
#define EPSV 1e-6f

typedef __attribute__((ext_vector_type(16))) __bf16 v16bf;
typedef __attribute__((ext_vector_type(8)))  float  v8f;

union BF16Frag { v16bf v; uint4 q[2]; };

__device__ __forceinline__ unsigned short f2bf(float f) {
  unsigned int u = __float_as_uint(f);
  unsigned int r = ((u >> 16) & 1u) + 0x7FFFu;
  return (unsigned short)((u + r) >> 16);
}
__device__ __forceinline__ float bf2f(unsigned short h) {
  return __uint_as_float(((unsigned int)h) << 16);
}
// Load a 16-element bf16 fragment as two contiguous 16-byte chunks.
__device__ __forceinline__ v16bf ldfrag(const unsigned short* p0, const unsigned short* p1) {
  BF16Frag f;
  f.q[0] = *(const uint4*)p0;
  f.q[1] = *(const uint4*)p1;
  return f.v;
}

// ---------------------------------------------------------------------------
// GEMM: C[M,N] = A[M,K] (bf16 row-major) * B[K,N], B supplied transposed as
// BT[N,K] bf16. One wave computes 64x64 (16 WMMA per 8 fragment loads);
// block = 8 waves (2 M x 4 N) = 128x256 tile.
// ---------------------------------------------------------------------------
template<bool OUT_BF16>
__global__ __launch_bounds__(256) void gemm_bf16_kernel(
    const unsigned short* __restrict__ A,
    const unsigned short* __restrict__ BT,
    void* __restrict__ Cout, int M, int N, int K)
{
  const int lane = threadIdx.x & 31;
  const int wave = threadIdx.x >> 5;
  const int l16  = lane & 15;
  const int hsel = lane >> 4;
  const int row0 = blockIdx.y * 128 + (wave >> 2) * 64;
  const int col0 = blockIdx.x * 256 + (wave & 3) * 64;

  v8f acc[4][4];
#pragma unroll
  for (int i = 0; i < 4; i++)
#pragma unroll
    for (int j = 0; j < 4; j++) acc[i][j] = {};

  for (int k0 = 0; k0 < K; k0 += 32) {
    v16bf af[4], bfr[4];
#pragma unroll
    for (int mt = 0; mt < 4; mt++) {
      const unsigned short* p = A + (size_t)(row0 + mt*16 + l16) * K + k0 + hsel*8;
      af[mt] = ldfrag(p, p + 16);          // A layout: K chunks [kb..kb+7],[kb+16..kb+23]
    }
#pragma unroll
    for (int nt = 0; nt < 4; nt++) {
      const unsigned short* p = BT + (size_t)(col0 + nt*16 + l16) * K + k0 + hsel*8;
      bfr[nt] = ldfrag(p, p + 16);         // B layout: K chunks [kb..kb+7],[kb+16..kb+23]
    }
#pragma unroll
    for (int mt = 0; mt < 4; mt++)
#pragma unroll
      for (int nt = 0; nt < 4; nt++)
        acc[mt][nt] = __builtin_amdgcn_wmma_f32_16x16x32_bf16(
            false, af[mt], false, bfr[nt], (short)0, acc[mt][nt], false, false);
  }

#pragma unroll
  for (int mt = 0; mt < 4; mt++)
#pragma unroll
    for (int nt = 0; nt < 4; nt++)
#pragma unroll
      for (int r = 0; r < 8; r++) {
        int row = row0 + mt*16 + r + hsel*8;
        int col = col0 + nt*16 + l16;
        if constexpr (OUT_BF16)
          ((unsigned short*)Cout)[(size_t)row * N + col] = f2bf(acc[mt][nt][r]);
        else
          ((float*)Cout)[(size_t)row * N + col] = acc[mt][nt][r];
      }
}

// ---------------------------------------------------------------------------
// Converters
// ---------------------------------------------------------------------------
__global__ __launch_bounds__(256) void cvt_f32_bf16_kernel(
    const float* __restrict__ in, unsigned short* __restrict__ out, int n4)
{
  int i = blockIdx.x * 256 + threadIdx.x;
  if (i < n4) {
    float4 v = ((const float4*)in)[i];
    unsigned short r[4] = { f2bf(v.x), f2bf(v.y), f2bf(v.z), f2bf(v.w) };
    ((uint2*)out)[i] = *(const uint2*)r;
  }
}

// in: rows x cols f32; out: cols x rows bf16 (LDS-tiled 32x32 transpose)
__global__ __launch_bounds__(256) void transpose_cvt_kernel(
    const float* __restrict__ in, unsigned short* __restrict__ out, int rows, int cols)
{
  __shared__ float tile[32][33];
  const int c0 = blockIdx.x * 32;
  const int r0 = blockIdx.y * 32;
  const int tx = threadIdx.x & 31;
  const int ty = threadIdx.x >> 5;
#pragma unroll
  for (int j = 0; j < 4; j++)
    tile[ty + 8*j][tx] = in[(size_t)(r0 + ty + 8*j) * cols + c0 + tx];
  __syncthreads();
#pragma unroll
  for (int j = 0; j < 4; j++)
    out[(size_t)(c0 + ty + 8*j) * rows + r0 + tx] = f2bf(tile[tx][ty + 8*j]);
}

// V slice of bf16 QKV -> Vt[b][kv][d][s], LDS-tiled 32x32 transpose.
__global__ __launch_bounds__(256) void vtrans_kernel(
    const unsigned short* __restrict__ qkv, unsigned short* __restrict__ Vt)
{
  __shared__ unsigned short tile[32][34];
  const int bk = blockIdx.z;                   // b*NKV + kv
  const int s0 = blockIdx.y * 32;
  const int d0 = blockIdx.x * 32;
  const int tx = threadIdx.x & 31;
  const int ty = threadIdx.x >> 5;
  const unsigned short* src = qkv + QSZ + KVSZ
      + (size_t)(bk % NKV) * HD + (size_t)(bk / NKV) * SS * OQKV;
#pragma unroll
  for (int j = 0; j < 4; j++)
    tile[ty + 8*j][tx] = src[(size_t)(s0 + ty + 8*j) * OQKV + d0 + tx];
  __syncthreads();
  unsigned short* dst = Vt + (size_t)bk * HD * SS;
#pragma unroll
  for (int j = 0; j < 4; j++)
    dst[(size_t)(d0 + ty + 8*j) * SS + s0 + tx] = tile[tx][ty + 8*j];
}

// ---------------------------------------------------------------------------
// RMSNorm + RoPE: one wave per (token, q-or-k head). 4 elems/lane over HD=128.
// ---------------------------------------------------------------------------
__global__ __launch_bounds__(256) void normrope_kernel(
    const unsigned short* __restrict__ qkv, const int* __restrict__ positions,
    const float* __restrict__ qw, const float* __restrict__ kw,
    unsigned short* __restrict__ Qb, unsigned short* __restrict__ Kb)
{
  const int lane = threadIdx.x & 31;
  const int wave = threadIdx.x >> 5;
  const int gid  = blockIdx.x * 8 + wave;
  const int t  = gid / (NHD + NKV);
  const int hh = gid % (NHD + NKV);
  const int b = t / SS, s = t % SS;
  const bool isq = hh < NHD;
  const int h = isq ? hh : hh - NHD;
  const float* w = isq ? qw : kw;
  const size_t src = (size_t)t * OQKV + (isq ? h * HD : QSZ + h * HD);

  float x[4];
  float ss2 = 0.f;
#pragma unroll
  for (int j = 0; j < 4; j++) { x[j] = bf2f(qkv[src + lane + 32*j]); ss2 += x[j]*x[j]; }
#pragma unroll
  for (int off = 16; off >= 1; off >>= 1) ss2 += __shfl_xor(ss2, off);
  float rstd = rsqrtf(ss2 * (1.0f / HD) + EPSV);
#pragma unroll
  for (int j = 0; j < 4; j++) x[j] = x[j] * rstd * w[lane + 32*j];

  float pos = (float)positions[t];
  float out[4];
#pragma unroll
  for (int p2 = 0; p2 < 2; p2++) {
    int i = lane + 32 * p2;                 // rope frequency index (0..63)
    float inv = __expf(-9.210340371976184f * ((float)i * (1.0f / 64.0f))); // 10000^(-i/64)
    float ang = pos * inv;
    float sn, cs; __sincosf(ang, &sn, &cs);
    float x1 = x[p2], x2 = x[p2 + 2];       // d=i pairs with d=i+64
    out[p2]     = x1 * cs - x2 * sn;
    out[p2 + 2] = x2 * cs + x1 * sn;
  }
  unsigned short* dst = isq ? Qb : Kb;
  size_t base = ((size_t)(b * (isq ? NHD : NKV) + h) * SS + s) * HD;
#pragma unroll
  for (int j = 0; j < 4; j++) dst[base + lane + 32*j] = f2bf(out[j]);
}

// ---------------------------------------------------------------------------
// Flash attention: one wave per (b, h, 16-query tile). Online softmax in f32.
// All K and V fragments for a 32-key tile are issued before the QK WMMAs so
// the V loads' latency is hidden under the softmax VALU work.
// ---------------------------------------------------------------------------
__global__ __launch_bounds__(256) void attn_kernel(
    const unsigned short* __restrict__ Qb,
    const unsigned short* __restrict__ Kb,
    const unsigned short* __restrict__ Vt,
    unsigned short* __restrict__ Ctx)
{
  __shared__ unsigned short pl[8][16][32];   // per-wave P tile (bf16), 8 KB
  const int lane = threadIdx.x & 31;
  const int wave = threadIdx.x >> 5;
  const int l16  = lane & 15;
  const int hsel = lane >> 4;
  const int task = blockIdx.x * 8 + wave;    // exact fill: B*NH*(S/16) tasks
  const int qt = task % (SS / 16);
  const int bh = task / (SS / 16);
  const int b = bh / NHD, h = bh % NHD;
  const int kv = h / (NHD / NKV);

  const unsigned short* Qbase = Qb + ((size_t)bh * SS + qt * 16) * HD;
  const unsigned short* Kbase = Kb + (size_t)(b * NKV + kv) * SS * HD;
  const unsigned short* Vbase = Vt + (size_t)(b * NKV + kv) * HD * SS;

  v16bf qf[4];                               // whole Q row tile resident in VGPRs
#pragma unroll
  for (int d = 0; d < 4; d++) {
    const unsigned short* p = Qbase + (size_t)l16 * HD + d * 32 + hsel * 8;
    qf[d] = ldfrag(p, p + 16);
  }

  v8f oacc[8];
#pragma unroll
  for (int i = 0; i < 8; i++) oacc[i] = {};
  float mrow[8], lrow[8];
#pragma unroll
  for (int r = 0; r < 8; r++) { mrow[r] = -1e30f; lrow[r] = 0.f; }

  for (int kk = 0; kk < SS; kk += 32) {
    if (kk + 32 < SS) {
      __builtin_prefetch(Kbase + (size_t)(kk + 32 + lane) * HD, 0, 0);
      __builtin_prefetch(Vbase + (size_t)(lane * 4) * SS + kk + 32, 0, 0);
    }

    // Issue all K fragments, then all V fragments (one long load clause).
    v16bf kf[8];
#pragma unroll
    for (int d = 0; d < 4; d++) {
      const unsigned short* p0 = Kbase + (size_t)(kk + l16) * HD + d * 32 + hsel * 8;
      const unsigned short* p1 = Kbase + (size_t)(kk + 16 + l16) * HD + d * 32 + hsel * 8;
      kf[d]     = ldfrag(p0, p0 + 16);
      kf[4 + d] = ldfrag(p1, p1 + 16);
    }
    v16bf vf[8];
#pragma unroll
    for (int nt = 0; nt < 8; nt++) {
      const unsigned short* vp = Vbase + (size_t)(nt * 16 + l16) * SS + kk + hsel * 8;
      vf[nt] = ldfrag(vp, vp + 16);
    }

    // scores for 32 keys: two 16x16 f32 fragments
    v8f s0 = {}, s1 = {};
#pragma unroll
    for (int d = 0; d < 4; d++) {
      s0 = __builtin_amdgcn_wmma_f32_16x16x32_bf16(false, qf[d], false, kf[d],     (short)0, s0, false, false);
      s1 = __builtin_amdgcn_wmma_f32_16x16x32_bf16(false, qf[d], false, kf[4 + d], (short)0, s1, false, false);
    }

    // online softmax: per-row max/sum via half-wave shuffles (rows live per-half)
    float mnew[8], alpha[8];
#pragma unroll
    for (int r = 0; r < 8; r++) {
      float a = s0[r] * SCL, c = s1[r] * SCL;
      s0[r] = a; s1[r] = c;
      float t0 = fmaxf(a, c);
      t0 = fmaxf(t0, __shfl_xor(t0, 1));
      t0 = fmaxf(t0, __shfl_xor(t0, 2));
      t0 = fmaxf(t0, __shfl_xor(t0, 4));
      t0 = fmaxf(t0, __shfl_xor(t0, 8));
      mnew[r]  = fmaxf(mrow[r], t0);
      alpha[r] = __expf(mrow[r] - mnew[r]);
      mrow[r]  = mnew[r];
    }
#pragma unroll
    for (int r = 0; r < 8; r++) {
      float p0 = __expf(s0[r] - mnew[r]);
      float p1 = __expf(s1[r] - mnew[r]);
      s0[r] = p0; s1[r] = p1;
      float rs = p0 + p1;
      rs += __shfl_xor(rs, 1);
      rs += __shfl_xor(rs, 2);
      rs += __shfl_xor(rs, 4);
      rs += __shfl_xor(rs, 8);
      lrow[r] = lrow[r] * alpha[r] + rs;
    }
#pragma unroll
    for (int nt = 0; nt < 8; nt++)
#pragma unroll
      for (int r = 0; r < 8; r++) oacc[nt][r] *= alpha[r];

    // C-layout -> A-layout for P via LDS (bf16), same-wave ordering via DScnt
#pragma unroll
    for (int r = 0; r < 8; r++) {
      pl[wave][r + hsel * 8][l16]      = f2bf(s0[r]);
      pl[wave][r + hsel * 8][16 + l16] = f2bf(s1[r]);
    }
    asm volatile("s_wait_dscnt 0" ::: "memory");
    const unsigned short* pp = &pl[wave][l16][hsel * 8];
    v16bf pf = ldfrag(pp, pp + 16);

    // O += P * V  (V fragments already resident)
#pragma unroll
    for (int nt = 0; nt < 8; nt++)
      oacc[nt] = __builtin_amdgcn_wmma_f32_16x16x32_bf16(
          false, pf, false, vf[nt], (short)0, oacc[nt], false, false);
  }

#pragma unroll
  for (int r = 0; r < 8; r++) lrow[r] = 1.0f / lrow[r];
#pragma unroll
  for (int nt = 0; nt < 8; nt++)
#pragma unroll
    for (int r = 0; r < 8; r++) {
      int srow = qt * 16 + r + hsel * 8;
      size_t t = (size_t)b * SS + srow;
      Ctx[t * QSZ + (size_t)h * HD + nt * 16 + l16] = f2bf(oacc[nt][r] * lrow[r]);
    }
}

// ---------------------------------------------------------------------------
extern "C" void kernel_launch(void* const* d_in, const int* in_sizes, int n_in,
                              void* d_out, int out_size, void* d_ws, size_t ws_size,
                              hipStream_t stream) {
  (void)in_sizes; (void)n_in; (void)out_size; (void)ws_size;
  const int*   positions = (const int*)d_in[0];
  const float* hidden    = (const float*)d_in[1];
  const float* w_qkv     = (const float*)d_in[2];
  const float* q_norm_w  = (const float*)d_in[3];
  const float* k_norm_w  = (const float*)d_in[4];
  const float* w_o       = (const float*)d_in[5];
  float* out = (float*)d_out;

  char* ws = (char*)d_ws;
  unsigned short* Xb    = (unsigned short*)ws; ws += (size_t)NTOK * HH   * 2; // 32 MB
  unsigned short* WqkvT = (unsigned short*)ws; ws += (size_t)OQKV * HH   * 2; // 16 MB
  unsigned short* WoT   = (unsigned short*)ws; ws += (size_t)HH   * QSZ  * 2; //  8 MB
  unsigned short* QKVb  = (unsigned short*)ws; ws += (size_t)NTOK * OQKV * 2; // 64 MB
  unsigned short* Qb    = (unsigned short*)ws; ws += (size_t)NTOK * QSZ  * 2; // 32 MB
  unsigned short* Kb    = (unsigned short*)ws; ws += (size_t)NTOK * KVSZ * 2; // 16 MB
  unsigned short* Vt    = (unsigned short*)ws; ws += (size_t)NTOK * KVSZ * 2; // 16 MB
  unsigned short* Ctx   = (unsigned short*)ws;                                // 32 MB

  cvt_f32_bf16_kernel<<<(NTOK * HH / 4) / 256, 256, 0, stream>>>(hidden, Xb, NTOK * HH / 4);
  transpose_cvt_kernel<<<dim3(OQKV / 32, HH / 32), 256, 0, stream>>>(w_qkv, WqkvT, HH, OQKV);
  transpose_cvt_kernel<<<dim3(HH / 32, QSZ / 32), 256, 0, stream>>>(w_o, WoT, QSZ, HH);

  gemm_bf16_kernel<true><<<dim3(OQKV / 256, NTOK / 128), 256, 0, stream>>>(
      Xb, WqkvT, QKVb, NTOK, OQKV, HH);

  normrope_kernel<<<(NTOK * (NHD + NKV)) / 8, 256, 0, stream>>>(
      QKVb, positions, q_norm_w, k_norm_w, Qb, Kb);
  vtrans_kernel<<<dim3(HD / 32, SS / 32, BB * NKV), 256, 0, stream>>>(QKVb, Vt);

  attn_kernel<<<(BB * NHD * (SS / 16)) / 8, 256, 0, stream>>>(Qb, Kb, Vt, Ctx);

  gemm_bf16_kernel<false><<<dim3(HH / 256, NTOK / 128), 256, 0, stream>>>(
      Ctx, WoT, out, NTOK, HH, QSZ);
}
